// TransformerXL_86844238725503
// MI455X (gfx1250) — matrix operand order repaired and checked by reading, not compiled, not run on hardware
//
#include <hip/hip_runtime.h>

// ---------------- problem constants ----------------
#define Bv  4
#define Tv  512
#define Vv  32000
#define Dv  1024
#define Lv  12
#define Hv  16
#define Fv  4096
#define HDv 64
#define MROWS (Bv*Tv)   // 2048

// ---------------- WMMA GEMM tile config ----------------
#define BM 128
#define BN 128
#define BK 64
#define LDT 72   // LDS row stride in bf16 elements (144B, 16B-aligned, conflict-friendly)

typedef __bf16 bf16;
typedef bf16  v16bf __attribute__((ext_vector_type(16)));
typedef bf16  v8bf  __attribute__((ext_vector_type(8)));
typedef float v8f   __attribute__((ext_vector_type(8)));

__device__ __forceinline__ float gelu_tanh(float x) {
    float x3 = x * x * x;
    return 0.5f * x * (1.0f + tanhf(0.7978845608028654f * (x + 0.044715f * x3)));
}

// 64B global -> LDS async copy (4 x b128); IOFFSET applies to both addresses.
__device__ __forceinline__ void async_cp64(unsigned ldsoff, const bf16* gsrc) {
    asm volatile(
        "global_load_async_to_lds_b128 %0, %1, off\n\t"
        "global_load_async_to_lds_b128 %0, %1, off offset:16\n\t"
        "global_load_async_to_lds_b128 %0, %1, off offset:32\n\t"
        "global_load_async_to_lds_b128 %0, %1, off offset:48"
        :: "v"(ldsoff), "v"((unsigned long long)gsrc) : "memory");
}
__device__ __forceinline__ void wait_async0() {
    asm volatile("s_wait_asynccnt 0" ::: "memory");
}

// ============================================================================
// Batched bf16 GEMM: C[z] = act(A[z] @ B[z](^T) + bias) (+ residual)
// A: M x K bf16 (lda). B: bf16, [N,K] if transB (fast path, async) else [K,N].
// Outputs: fp32 C (ldc) and/or bf16 Cb (ldcb), both optional.
// Double-buffered LDS, tiles fed by GLOBAL_LOAD_ASYNC_TO_LDS_B128 (ASYNCcnt),
// compute via v_wmma_f32_16x16x32_bf16 (16 WMMAs per barrier pair per wave).
// ============================================================================
__global__ __launch_bounds__(256)
void gemm_bf16_kernel(const bf16* __restrict__ A, const bf16* __restrict__ Bm,
                      const float* __restrict__ bias, const float* __restrict__ Res,
                      float* __restrict__ C, bf16* __restrict__ Cb,
                      int M, int N, int K, int lda, int ldb, int ldc, int ldcb,
                      long long oAo, long long oAi, long long oBo, long long oBi,
                      long long oCo, long long oCi, int innerCnt,
                      int transB, int act)
{
    __shared__ bf16 As[2][BM * LDT];
    __shared__ bf16 Bs[2][BN * LDT];

    const int z  = blockIdx.z;
    const int zo = z / innerCnt;
    const int zi = z - zo * innerCnt;
    const bf16* Ab = A  + zo * oAo + zi * oAi;
    const bf16* Bb = Bm + zo * oBo + zi * oBi;

    const int m0 = blockIdx.y * BM;
    const int n0 = blockIdx.x * BN;
    const int tid  = threadIdx.x;
    const int lane = tid & 31;
    const int wave = tid >> 5;
    const int wm = wave & 3;        // 0..3 -> 32-row group
    const int wn = wave >> 2;       // 0..1 -> 64-col group
    const int lo = lane & 15;
    const int hi = lane >> 4;

    // cooperative loader indices: 2 threads per row, 32 bf16 (64B) each
    const int arow  = tid >> 1;
    const int ahalf = (tid & 1) << 5;     // 0 or 32 elements
    // transB=0 transpose loader: 4 threads per k-row, 32 n-elements each
    const int bk  = tid >> 2;             // 0..63
    const int bnq = (tid & 3) << 5;       // 0,32,64,96

    const unsigned asbase = (unsigned)(unsigned long long)(void*)&As[0][0];
    const unsigned bsbase = (unsigned)(unsigned long long)(void*)&Bs[0][0];

    const bool edgeM = (m0 + BM > M);
    const bool edgeN = (n0 + BN > N);

    auto loadA = [&](int buf, int kt) {
        int gr = m0 + arow;
        if (!edgeM) {
            async_cp64(asbase + (unsigned)(buf * (BM * LDT) + arow * LDT + ahalf) * 2u,
                       Ab + (long long)gr * lda + kt + ahalf);
        } else {
            uint4* dst = (uint4*)&As[buf][arow * LDT + ahalf];
            if (gr < M) {
                const uint4* src = (const uint4*)(Ab + (long long)gr * lda + kt + ahalf);
                #pragma unroll
                for (int j = 0; j < 4; ++j) dst[j] = src[j];
            } else {
                uint4 zz; zz.x = zz.y = zz.z = zz.w = 0u;
                #pragma unroll
                for (int j = 0; j < 4; ++j) dst[j] = zz;
            }
        }
    };
    auto loadB = [&](int buf, int kt) {
        if (transB) {                       // B stored [N,K]: contiguous copy
            int gn = n0 + arow;
            if (!edgeN) {
                async_cp64(bsbase + (unsigned)(buf * (BN * LDT) + arow * LDT + ahalf) * 2u,
                           Bb + (long long)gn * ldb + kt + ahalf);
            } else {
                uint4* dst = (uint4*)&Bs[buf][arow * LDT + ahalf];
                if (gn < N) {
                    const uint4* src = (const uint4*)(Bb + (long long)gn * ldb + kt + ahalf);
                    #pragma unroll
                    for (int j = 0; j < 4; ++j) dst[j] = src[j];
                } else {
                    uint4 zz; zz.x = zz.y = zz.z = zz.w = 0u;
                    #pragma unroll
                    for (int j = 0; j < 4; ++j) dst[j] = zz;
                }
            }
        } else {                            // B stored [K,N]: transpose via LDS
            const bf16* src = Bb + (long long)(kt + bk) * ldb + n0 + bnq;
            #pragma unroll
            for (int j = 0; j < 32; ++j) {
                bf16 v = (n0 + bnq + j < N) ? src[j] : (bf16)0.0f;
                Bs[buf][(bnq + j) * LDT + bk] = v;
            }
        }
    };

    v8f acc[2][4];
    #pragma unroll
    for (int i = 0; i < 2; ++i)
        #pragma unroll
        for (int j = 0; j < 4; ++j) acc[i][j] = {};

    loadA(0, 0); loadB(0, 0);
    wait_async0();
    __syncthreads();

    int cur = 0;
    for (int kt = 0; kt < K; kt += BK) {
        int nxt = cur ^ 1;
        if (kt + BK < K) { loadA(nxt, kt + BK); loadB(nxt, kt + BK); }

        union U16 { v16bf v; v8bf h[2]; };
        #pragma unroll
        for (int ks = 0; ks < 2; ++ks) {
            const int kk = ks * 32;
            U16 afr[2], bfr[4];
            #pragma unroll
            for (int i = 0; i < 2; ++i) {
                const bf16* ap = &As[cur][(wm * 32 + i * 16 + lo) * LDT + kk];
                afr[i].h[0] = *(const v8bf*)&ap[hi * 8];          // K = kk+8*hi..
                afr[i].h[1] = *(const v8bf*)&ap[16 + hi * 8];     // K = kk+16+8*hi..
            }
            #pragma unroll
            for (int j = 0; j < 4; ++j) {
                const bf16* bp = &Bs[cur][(wn * 64 + j * 16 + lo) * LDT + kk];
                bfr[j].h[0] = *(const v8bf*)&bp[hi * 8];          // K = kk+8*hi..
                bfr[j].h[1] = *(const v8bf*)&bp[16 + hi * 8];
            }
            #pragma unroll
            for (int i = 0; i < 2; ++i)
                #pragma unroll
                for (int j = 0; j < 4; ++j)
                    acc[i][j] = __builtin_amdgcn_wmma_f32_16x16x32_bf16(
                        false, afr[i].v, false, bfr[j].v, (short)0, acc[i][j], false, false);
        }
        wait_async0();
        __syncthreads();
        cur = nxt;
    }

    // ---- epilogue: bias -> activation -> residual -> fp32 and/or bf16 store ----
    float*       Cp  = C  ? (C  + zo * oCo + zi * oCi) : nullptr;
    bf16*        Cbp = Cb ? (Cb + zo * oCo + zi * oCi) : nullptr;
    const float* Rp  = Res ? (Res + zo * oCo + zi * oCi) : nullptr;
    #pragma unroll
    for (int i = 0; i < 2; ++i) {
        int rb = m0 + wm * 32 + i * 16 + hi * 8;
        #pragma unroll
        for (int j = 0; j < 4; ++j) {
            int cb = n0 + wn * 64 + j * 16 + lo;
            float bval = (bias && cb < N) ? bias[cb] : 0.0f;
            #pragma unroll
            for (int r = 0; r < 8; ++r) {
                int rr = rb + r;
                if (rr < M && cb < N) {
                    float val = acc[i][j][r] + bval;
                    if (act == 1) val = gelu_tanh(val);
                    if (Rp) val += Rp[(long long)rr * ldc + cb];
                    if (Cp)  Cp[(long long)rr * ldc + cb] = val;
                    if (Cbp) Cbp[(long long)rr * ldcb + cb] = (bf16)val;
                }
            }
        }
    }
}

// ---- fp32 -> bf16 straight convert ----
__global__ __launch_bounds__(256)
void cvt_kernel(const float* __restrict__ in, bf16* __restrict__ outp, int n)
{
    int i = blockIdx.x * 256 + threadIdx.x;
    if (i < n) outp[i] = (bf16)in[i];
}

// ---- fp32 [R,C] -> bf16 [C,R] transposed convert (weight repack to [N,K]) ----
__global__ __launch_bounds__(256)
void cvt_t_kernel(const float* __restrict__ in, bf16* __restrict__ outp, int R, int C)
{
    __shared__ float t[32][33];
    int c0 = blockIdx.x * 32, r0 = blockIdx.y * 32;
    int tx = threadIdx.x & 31, ty = threadIdx.x >> 5;   // ty 0..7
    #pragma unroll
    for (int u = 0; u < 4; ++u) {
        int r = r0 + ty + u * 8;
        t[ty + u * 8][tx] = (r < R && c0 + tx < C) ? in[(size_t)r * C + c0 + tx] : 0.0f;
    }
    __syncthreads();
    #pragma unroll
    for (int u = 0; u < 4; ++u) {
        int c = c0 + ty + u * 8;
        if (c < C && r0 + tx < R)
            outp[(size_t)c * R + r0 + tx] = (bf16)t[tx][ty + u * 8];
    }
}

// ---- adaptive input embedding (gather + up-projection), scaled by sqrt(D) ----
__global__ __launch_bounds__(256)
void embed_kernel(const int* __restrict__ x, const float* __restrict__ e0,
                  const float* __restrict__ e1, const float* __restrict__ e2,
                  const float* __restrict__ p1, const float* __restrict__ p2,
                  float* __restrict__ h)
{
    __shared__ float ebuf[256];
    int tok = blockIdx.x;
    int id  = x[tok];
    float* hr = h + (size_t)tok * Dv;
    if (id < 2000) {
        const float* er = e0 + (size_t)id * Dv;
        for (int d = threadIdx.x; d < Dv; d += 256) hr[d] = er[d] * 32.0f;
    } else if (id < 8000) {
        int r = id - 2000;
        ebuf[threadIdx.x] = e1[(size_t)r * 256 + threadIdx.x];
        __syncthreads();
        for (int d = threadIdx.x; d < Dv; d += 256) {
            float acc = 0.0f;
            for (int j = 0; j < 256; ++j) acc += ebuf[j] * p1[(size_t)j * Dv + d];
            hr[d] = acc * 32.0f;
        }
    } else {
        int r = id - 8000;
        if (threadIdx.x < 64) ebuf[threadIdx.x] = e2[(size_t)r * 64 + threadIdx.x];
        __syncthreads();
        for (int d = threadIdx.x; d < Dv; d += 256) {
            float acc = 0.0f;
            for (int j = 0; j < 64; ++j) acc += ebuf[j] * p2[(size_t)j * Dv + d];
            hr[d] = acc * 32.0f;
        }
    }
}

// ---- sinusoid position embedding (bf16, TXL convention pos = T-1..0) ----
__global__ __launch_bounds__(256)
void pemb_kernel(bf16* __restrict__ pe)
{
    int row = blockIdx.x;
    float pos = (float)(Tv - 1 - row);
    for (int c = threadIdx.x; c < Dv / 2; c += 256) {
        float invf = powf(10000.0f, -(2.0f * (float)c) / (float)Dv);
        float ang  = pos * invf;
        pe[(size_t)row * Dv + c]          = (bf16)sinf(ang);
        pe[(size_t)row * Dv + Dv / 2 + c] = (bf16)cosf(ang);
    }
}

// ---- LayerNorm: fp32 in, bf16 out; one 256-thread block per row of D=1024 ----
__global__ __launch_bounds__(256)
void layernorm_kernel(const float* __restrict__ x, const float* __restrict__ s,
                      const float* __restrict__ b, bf16* __restrict__ y)
{
    int row = blockIdx.x;
    const float* xr = x + (size_t)row * Dv;
    float v0[4], sum = 0.0f, sq = 0.0f;
    #pragma unroll
    for (int u = 0; u < 4; ++u) {
        float f = xr[threadIdx.x + u * 256];
        v0[u] = f; sum += f; sq += f * f;
    }
    #pragma unroll
    for (int o = 16; o > 0; o >>= 1) {
        sum += __shfl_xor(sum, o, 32);
        sq  += __shfl_xor(sq,  o, 32);
    }
    __shared__ float rs[8], rq[8];
    int lane = threadIdx.x & 31, w = threadIdx.x >> 5;
    if (lane == 0) { rs[w] = sum; rq[w] = sq; }
    __syncthreads();
    sum = 0.0f; sq = 0.0f;
    #pragma unroll
    for (int i = 0; i < 8; ++i) { sum += rs[i]; sq += rq[i]; }
    float mean = sum * (1.0f / Dv);
    float var  = sq * (1.0f / Dv) - mean * mean;
    float inv  = rsqrtf(var + 1e-5f);
    bf16* yr = y + (size_t)row * Dv;
    #pragma unroll
    for (int u = 0; u < 4; ++u) {
        int c = threadIdx.x + u * 256;
        yr[c] = (bf16)((v0[u] - mean) * inv * s[c] + b[c]);
    }
}

// ---- q_rw = q + r_w_bias, q_rr = q + r_r_bias -> bf16 ----
__global__ __launch_bounds__(256)
void qbias_kernel(const float* __restrict__ q, const float* __restrict__ rwb,
                  const float* __restrict__ rrb, bf16* __restrict__ qrw,
                  bf16* __restrict__ qrr, int n)
{
    int i = blockIdx.x * 256 + threadIdx.x;
    if (i < n) {
        int c = i & (Dv - 1);
        float v = q[i];
        qrw[i] = (bf16)(v + rwb[c]);
        qrr[i] = (bf16)(v + rrb[c]);
    }
}

// ---- fused rel_shift + causal mask + scale + softmax -> bf16 probs ----
// Under causal mask: rel_shift(rel)[t,s] = rel[t, s + T-1 - t]  (s <= t)
__global__ __launch_bounds__(256)
void relshift_softmax_kernel(const float* __restrict__ content,
                             const float* __restrict__ rel,
                             bf16* __restrict__ attn, float scale)
{
    int row  = blockIdx.x * 8 + (threadIdx.x >> 5);   // 0 .. B*H*T-1
    int lane = threadIdx.x & 31;
    int t  = row & (Tv - 1);
    int bh = row >> 9;
    const float* crow = content + (size_t)bh * Tv * Tv + (size_t)t * Tv;
    const float* rrow = rel     + (size_t)bh * Tv * Tv + (size_t)t * Tv;
    float vals[16];
    float mx = -3.0e38f;
    #pragma unroll
    for (int j = 0; j < 16; ++j) {
        int s2 = lane + 32 * j;
        float v;
        if (s2 <= t) v = (crow[s2] + rrow[s2 + (Tv - 1) - t]) * scale;
        else         v = -3.0e38f;
        vals[j] = v;
        mx = fmaxf(mx, v);
    }
    #pragma unroll
    for (int o = 16; o > 0; o >>= 1) mx = fmaxf(mx, __shfl_xor(mx, o, 32));
    float sum = 0.0f;
    #pragma unroll
    for (int j = 0; j < 16; ++j) {
        float e = (vals[j] > -1.0e37f) ? __expf(vals[j] - mx) : 0.0f;
        vals[j] = e; sum += e;
    }
    #pragma unroll
    for (int o = 16; o > 0; o >>= 1) sum += __shfl_xor(sum, o, 32);
    float inv = 1.0f / sum;
    bf16* wrow = attn + (size_t)bh * Tv * Tv + (size_t)t * Tv;
    #pragma unroll
    for (int j = 0; j < 16; ++j) wrow[lane + 32 * j] = (bf16)(vals[j] * inv);
}

// ============================================================================
// host side
// ============================================================================
static inline void gemm(hipStream_t st, const bf16* A, const bf16* B,
                        const float* bias, const float* res,
                        float* C, bf16* Cb,
                        int M, int N, int K, int lda, int ldb, int ldc, int ldcb,
                        int transB, int act, int batch = 1,
                        long long aO = 0, long long aI = 0,
                        long long bO = 0, long long bI = 0,
                        long long cO = 0, long long cI = 0, int inner = 1)
{
    dim3 g((N + BN - 1) / BN, (M + BM - 1) / BM, batch);
    hipLaunchKernelGGL(gemm_bf16_kernel, g, dim3(256), 0, st,
                       A, B, bias, res, C, Cb, M, N, K, lda, ldb, ldc, ldcb,
                       aO, aI, bO, bI, cO, cI, inner, transB, act);
}
static inline void cvt(hipStream_t st, const float* in, bf16* out, size_t n)
{
    hipLaunchKernelGGL(cvt_kernel, dim3((unsigned)((n + 255) / 256)), dim3(256), 0, st,
                       in, out, (int)n);
}
static inline void cvt_t(hipStream_t st, const float* in, bf16* out, int R, int C)
{
    dim3 g((C + 31) / 32, (R + 31) / 32);
    hipLaunchKernelGGL(cvt_t_kernel, g, dim3(256), 0, st, in, out, R, C);
}

extern "C" void kernel_launch(void* const* d_in, const int* in_sizes, int n_in,
                              void* d_out, int out_size, void* d_ws, size_t ws_size,
                              hipStream_t stream)
{
    const int*   x      = (const int*)  d_in[0];
    const float* emb_0  = (const float*)d_in[1];
    const float* emb_1  = (const float*)d_in[2];
    const float* emb_2  = (const float*)d_in[3];
    const float* proj_1 = (const float*)d_in[4];
    const float* proj_2 = (const float*)d_in[5];
    const float* outb   = (const float*)d_in[6];
    const float* rwb    = (const float*)d_in[7];
    const float* rrb    = (const float*)d_in[8];
    const float* ln1_s  = (const float*)d_in[9];
    const float* ln1_b  = (const float*)d_in[10];
    const float* Wq     = (const float*)d_in[11];
    const float* bq     = (const float*)d_in[12];
    const float* Wk     = (const float*)d_in[13];
    const float* bk     = (const float*)d_in[14];
    const float* Wv     = (const float*)d_in[15];
    const float* bvb    = (const float*)d_in[16];
    const float* Wr     = (const float*)d_in[17];
    const float* Wo     = (const float*)d_in[18];
    const float* bo     = (const float*)d_in[19];
    const float* ln2_s  = (const float*)d_in[20];
    const float* ln2_b  = (const float*)d_in[21];
    const float* W1     = (const float*)d_in[22];
    const float* b1     = (const float*)d_in[23];
    const float* W2     = (const float*)d_in[24];
    const float* b2     = (const float*)d_in[25];
    float* out = (float*)d_out;

    // ---- workspace: fp32 region then bf16 region ----
    float* F = (float*)d_ws;
    size_t fo = 0;
    auto falloc = [&](size_t n) { float* p = F + fo; fo += (n + 63) & ~(size_t)63; return p; };
    float* h       = falloc((size_t)MROWS * Dv);
    float* qb      = falloc((size_t)MROWS * Dv);
    float* content = falloc((size_t)Bv * Hv * Tv * Tv);
    float* rel     = falloc((size_t)Bv * Hv * Tv * Tv);

    bf16* BB = (bf16*)(F + fo);
    size_t bo2 = 0;
    auto balloc = [&](size_t n) { bf16* p = BB + bo2; bo2 += (n + 63) & ~(size_t)63; return p; };
    bf16* pemb_bf = balloc((size_t)Tv * Dv);
    bf16* a_in_bf = balloc((size_t)MROWS * Dv);
    bf16* kb_bf   = balloc((size_t)MROWS * Dv);
    bf16* vb_bf   = balloc((size_t)MROWS * Dv);
    bf16* qrw_bf  = balloc((size_t)MROWS * Dv);
    bf16* qrr_bf  = balloc((size_t)MROWS * Dv);
    bf16* rk_bf   = balloc((size_t)Tv * Dv);
    bf16* attn_bf = balloc((size_t)Bv * Hv * Tv * Tv);
    bf16* ob_bf   = balloc((size_t)MROWS * Dv);
    bf16* mid_bf  = balloc((size_t)MROWS * Fv);
    bf16* h_bf    = balloc((size_t)MROWS * Dv);
    bf16* wfull   = balloc((size_t)Vv * Dv);
    bf16* p1t     = balloc((size_t)Dv * 256);
    bf16* p2t     = balloc((size_t)Dv * 64);
    bf16* e1_bf   = balloc((size_t)6000 * 256);
    bf16* e2_bf   = balloc((size_t)24000 * 64);
    // per-layer weight arena (repacked bf16, [N,K]); reused across layers
    bf16* wq_t = balloc((size_t)Dv * Dv);
    bf16* wk_t = balloc((size_t)Dv * Dv);
    bf16* wv_t = balloc((size_t)Dv * Dv);
    bf16* wr_t = balloc((size_t)Dv * Dv);
    bf16* wo_t = balloc((size_t)Dv * Dv);
    bf16* w1_t = balloc((size_t)Fv * Dv);   // [F,D]
    bf16* w2_t = balloc((size_t)Dv * Fv);   // [D,F]

    const long long sTD = (long long)Tv * Dv;
    const long long sTT = (long long)Tv * Tv;
    const long long sHT = (long long)Hv * sTT;

    // ---- one-time prep ----
    hipLaunchKernelGGL(embed_kernel, dim3(MROWS), dim3(256), 0, stream,
                       x, emb_0, emb_1, emb_2, proj_1, proj_2, h);
    hipLaunchKernelGGL(pemb_kernel, dim3(Tv), dim3(256), 0, stream, pemb_bf);
    cvt_t(stream, proj_1, p1t, 256, Dv);
    cvt_t(stream, proj_2, p2t, 64, Dv);
    cvt(stream, emb_1, e1_bf, (size_t)6000 * 256);
    cvt(stream, emb_2, e2_bf, (size_t)24000 * 64);
    cvt(stream, emb_0, wfull, (size_t)2000 * Dv);
    // W_full rows 2000.. : adaptive up-projections, bf16 out
    gemm(stream, e1_bf, p1t, nullptr, nullptr, nullptr, wfull + (size_t)2000 * Dv,
         6000, Dv, 256, 256, 256, 0, Dv, /*transB=*/1, 0);
    gemm(stream, e2_bf, p2t, nullptr, nullptr, nullptr, wfull + (size_t)8000 * Dv,
         24000, Dv, 64, 64, 64, 0, Dv, /*transB=*/1, 0);

    // ---- transformer layers ----
    for (int i = 0; i < Lv; ++i) {
        // repack this layer's weights fp32 -> bf16 [N,K]
        cvt_t(stream, Wq + (size_t)i * Dv * Dv, wq_t, Dv, Dv);
        cvt_t(stream, Wk + (size_t)i * Dv * Dv, wk_t, Dv, Dv);
        cvt_t(stream, Wv + (size_t)i * Dv * Dv, wv_t, Dv, Dv);
        cvt_t(stream, Wr + (size_t)i * Dv * Dv, wr_t, Dv, Dv);
        cvt_t(stream, Wo + (size_t)i * Dv * Dv, wo_t, Dv, Dv);
        cvt_t(stream, W1 + (size_t)i * Dv * Fv, w1_t, Dv, Fv);
        cvt_t(stream, W2 + (size_t)i * Fv * Dv, w2_t, Fv, Dv);

        hipLaunchKernelGGL(layernorm_kernel, dim3(MROWS), dim3(256), 0, stream,
                           h, ln1_s + (size_t)i * Dv, ln1_b + (size_t)i * Dv, a_in_bf);
        // q (fp32 for bias-combine), k, v, r_k (bf16)
        gemm(stream, a_in_bf, wq_t, bq + (size_t)i * Dv, nullptr, qb, nullptr,
             MROWS, Dv, Dv, Dv, Dv, Dv, 0, 1, 0);
        gemm(stream, a_in_bf, wk_t, bk + (size_t)i * Dv, nullptr, nullptr, kb_bf,
             MROWS, Dv, Dv, Dv, Dv, 0, Dv, 1, 0);
        gemm(stream, a_in_bf, wv_t, bvb + (size_t)i * Dv, nullptr, nullptr, vb_bf,
             MROWS, Dv, Dv, Dv, Dv, 0, Dv, 1, 0);
        gemm(stream, pemb_bf, wr_t, nullptr, nullptr, nullptr, rk_bf,
             Tv, Dv, Dv, Dv, Dv, 0, Dv, 1, 0);
        hipLaunchKernelGGL(qbias_kernel, dim3(MROWS * Dv / 256), dim3(256), 0, stream,
                           qb, rwb, rrb, qrw_bf, qrr_bf, MROWS * Dv);
        // content = (q+rwb) @ k^T  [batched (b,h)]
        gemm(stream, qrw_bf, kb_bf, nullptr, nullptr, content, nullptr,
             Tv, Tv, HDv, Dv, Dv, Tv, 0, 1, 0, Bv * Hv,
             sTD, HDv, sTD, HDv, sHT, sTT, Hv);
        // rel = (q+rrb) @ r_k^T    [batched; B shared across b]
        gemm(stream, qrr_bf, rk_bf, nullptr, nullptr, rel, nullptr,
             Tv, Tv, HDv, Dv, Dv, Tv, 0, 1, 0, Bv * Hv,
             sTD, HDv, 0, HDv, sHT, sTT, Hv);
        hipLaunchKernelGGL(relshift_softmax_kernel,
                           dim3(Bv * Hv * Tv / 8), dim3(256), 0, stream,
                           content, rel, attn_bf, 0.125f);
        // o = attn @ v   (B is [K,N]: transB=0 transpose path)
        gemm(stream, attn_bf, vb_bf, nullptr, nullptr, nullptr, ob_bf,
             Tv, HDv, Tv, Tv, Dv, 0, Dv, 0, 0, Bv * Hv,
             sHT, sTT, sTD, HDv, sTD, HDv, Hv);
        // h = h + o @ Wo + bo
        gemm(stream, ob_bf, wo_t, bo + (size_t)i * Dv, h, h, nullptr,
             MROWS, Dv, Dv, Dv, Dv, Dv, 0, 1, 0);
        // FFN (pre-norm, GELU)
        hipLaunchKernelGGL(layernorm_kernel, dim3(MROWS), dim3(256), 0, stream,
                           h, ln2_s + (size_t)i * Dv, ln2_b + (size_t)i * Dv, a_in_bf);
        gemm(stream, a_in_bf, w1_t, b1 + (size_t)i * Fv, nullptr, nullptr, mid_bf,
             MROWS, Fv, Dv, Dv, Dv, 0, Fv, 1, /*gelu=*/1);
        gemm(stream, mid_bf, w2_t, b2 + (size_t)i * Dv, h, h, nullptr,
             MROWS, Dv, Fv, Fv, Fv, Dv, 0, 1, 0);
    }

    // ---- final tied adaptive-softmax projection ----
    cvt(stream, h, h_bf, (size_t)MROWS * Dv);
    gemm(stream, h_bf, wfull, outb, nullptr, out, nullptr,
         MROWS, Vv, Dv, Dv, Dv, Vv, 0, /*transB=*/1, 0);
}
